// LinearRTU_Triton_8899172237348
// MI455X (gfx1250) — compile-verified, matrix operand order checked by reading, not run on hardware
//
#include <hip/hip_runtime.h>
#include <math.h>
#include <stdint.h>

typedef __bf16 bf16_t;
typedef __attribute__((ext_vector_type(16))) __bf16 v16bf;
typedef __attribute__((ext_vector_type(8)))  __bf16 v8bf;
typedef __attribute__((ext_vector_type(8)))  float  v8f;

// Problem sizes (fixed by the reference)
constexpr int Bb = 8, Tt = 2048, Dd = 1024, Rr = 256, Hh = 512;
constexpr int M  = Bb * Tt;           // 16384 rows
// GEMM tiling
constexpr int TM = 128, TN = 64, TK = 32;
constexpr int KT = Dd / TK;           // 32 k-tiles
constexpr int RS = 40;                // LDS row stride in bf16 (80B, bank-spread)

__device__ inline float silu_f(float x) { return x / (1.0f + expf(-x)); }

// Low 32 bits of a generic pointer to __shared__ data == LDS byte offset
// (ISA §10.2: LDS aperture address truncates to LDS_ADDR.U32).
__device__ inline uint32_t lds_addr32(const void* p) {
  return (uint32_t)(uintptr_t)p;
}

// ---------------------------------------------------------------- params
__global__ void params_kernel(const float* __restrict__ nu_log,
                              const float* __restrict__ theta_log,
                              float* __restrict__ g_out,
                              float* __restrict__ phi_out,
                              float* __restrict__ gamma_out) {
  int h = blockIdx.x * blockDim.x + threadIdx.x;
  if (h >= Hh) return;
  float r  = expf(-expf(nu_log[h]));
  float th = expf(theta_log[h]);
  g_out[h]   = r * cosf(th);
  phi_out[h] = r * sinf(th);
  float t = 1.0f - r * r;
  gamma_out[h] = sqrtf(t > 0.0f ? t : 0.0f);
}

// ---------------------------------------------------------------- x -> bf16
__global__ void convert_x_kernel(const float* __restrict__ x,
                                 bf16_t* __restrict__ xb) {
  int i = (blockIdx.x * blockDim.x + threadIdx.x) * 8;   // M*Dd divisible by 8
  float4 a = *(const float4*)(x + i);
  float4 b = *(const float4*)(x + i + 4);
  v8bf o;
  o[0] = (__bf16)a.x; o[1] = (__bf16)a.y; o[2] = (__bf16)a.z; o[3] = (__bf16)a.w;
  o[4] = (__bf16)b.x; o[5] = (__bf16)b.y; o[6] = (__bf16)b.z; o[7] = (__bf16)b.w;
  *(v8bf*)(xb + i) = o;
}

// ---------------------------------------------------------------- W = U@V, stored transposed [H][D] in bf16
__global__ void build_wt_kernel(const float* __restrict__ U1,
                                const float* __restrict__ U2,
                                const float* __restrict__ V1,
                                const float* __restrict__ V2,
                                bf16_t* __restrict__ W1t,
                                bf16_t* __restrict__ W2t) {
  int id = blockIdx.x * blockDim.x + threadIdx.x;   // Hh*Dd threads
  int d = id & (Dd - 1);
  int h = id >> 10;
  if (h >= Hh) return;
  float a1 = 0.0f, a2 = 0.0f;
  for (int rr = 0; rr < Rr; ++rr) {
    float u1 = U1[d * Rr + rr];
    float u2 = U2[d * Rr + rr];
    a1 = fmaf(u1, V1[rr * Hh + h], a1);
    a2 = fmaf(u2, V2[rr * Hh + h], a2);
  }
  W1t[h * Dd + d] = (bf16_t)a1;
  W2t[h * Dd + d] = (bf16_t)a2;
}

// ---------------------------------------------------------------- fused WMMA GEMM: bx=(x@W1)*gamma, by=(x@W2)*gamma
__device__ inline v16bf load_frag(const bf16_t* tile, int row, int kh) {
  // fragment = two contiguous 16B chunks per lane (ISA 16-bit A/B layout)
  v8bf lo = *(const v8bf*)(tile + row * RS + kh);
  v8bf hi = *(const v8bf*)(tile + row * RS + 16 + kh);
  return __builtin_shufflevector(lo, hi, 0, 1, 2, 3, 4, 5, 6, 7,
                                 8, 9, 10, 11, 12, 13, 14, 15);
}

__global__ void __launch_bounds__(256)
gemm_bxby_kernel(const bf16_t* __restrict__ xb,    // [M][Dd] bf16
                 const bf16_t* __restrict__ W1t,   // [Hh][Dd] bf16
                 const bf16_t* __restrict__ W2t,   // [Hh][Dd] bf16
                 const float*  __restrict__ gamma_,
                 float* __restrict__ bx,           // [M][Hh]
                 float* __restrict__ by) {
  // double-buffered tiles, filled by global_load_async_to_lds (ASYNCcnt)
  __shared__ __align__(16) bf16_t As [2][TM * RS];
  __shared__ __align__(16) bf16_t B1s[2][TN * RS];
  __shared__ __align__(16) bf16_t B2s[2][TN * RS];

  const int m0   = blockIdx.x * TM;
  const int n0   = blockIdx.y * TN;
  const int tid  = threadIdx.x;
  const int lane = tid & 31;
  const int wave = tid >> 5;                 // 0..7, each owns 16 M-rows

  // staging assignments: 2 threads per tile row, 32B (one b128 x2) each
  const int arow  = tid >> 1;                // 0..127
  const int ahalf = tid & 1;
  const int brow  = (tid & 127) >> 1;        // 0..63
  const int bhalf = tid & 1;
  const int bsel  = tid >> 7;                // 0 -> W1, 1 -> W2

  const bf16_t* aG = xb + (size_t)(m0 + arow) * Dd + ahalf * 16;
  const bf16_t* wG = (bsel ? W2t : W1t) + (size_t)(n0 + brow) * Dd + bhalf * 16;
  bf16_t* wL0 = (bsel ? &B2s[0][0] : &B1s[0][0]) + brow * RS + bhalf * 16;
  bf16_t* wL1 = (bsel ? &B2s[1][0] : &B1s[1][0]) + brow * RS + bhalf * 16;

  v8f acc1[4] = {};
  v8f acc2[4] = {};

  const int mrow = wave * 16 + (lane & 15);  // LDS A row for this lane
  const int kh   = (lane >> 4) * 8;          // element offset of lane's chunk pair

  // issue one K-tile's DMA into LDS buffer `buf` (no VGPR data movement)
  auto issue_tile = [&](int kc, int buf) {
    const bf16_t* ga = aG + kc * TK;
    const bf16_t* gw = wG + kc * TK;
    uint32_t la = lds_addr32(&As[buf][arow * RS + ahalf * 16]);
    uint32_t lw = lds_addr32(buf ? wL1 : wL0);
    asm volatile("global_load_async_to_lds_b128 %0, %1, off"
                 :: "v"(la), "v"(ga) : "memory");
    asm volatile("global_load_async_to_lds_b128 %0, %1, off offset:16"
                 :: "v"(la), "v"(ga) : "memory");
    asm volatile("global_load_async_to_lds_b128 %0, %1, off"
                 :: "v"(lw), "v"(gw) : "memory");
    asm volatile("global_load_async_to_lds_b128 %0, %1, off offset:16"
                 :: "v"(lw), "v"(gw) : "memory");
  };

  issue_tile(0, 0);

  for (int kc = 0; kc < KT; ++kc) {
    const int cur = kc & 1;
    // own DMA for tile kc has landed in LDS:
    asm volatile("s_wait_asynccnt 0x0" ::: "memory");
    // everyone's DMA landed; also guarantees all LDS reads of buffer cur^1
    // (from iteration kc-1) are drained before anyone refills it:
    __syncthreads();

    if (kc + 1 < KT) {
      issue_tile(kc + 1, cur ^ 1);           // overlap DMA with WMMA below
      if (kc + 2 < KT) {                     // warm L2 for the tile after
        __builtin_prefetch(aG + (kc + 2) * TK, 0, 1);
        __builtin_prefetch(wG + (kc + 2) * TK, 0, 1);
      }
    }

    const bf16_t* Ab  = &As [cur][0];
    const bf16_t* B1b = &B1s[cur][0];
    const bf16_t* B2b = &B2s[cur][0];

    v16bf afrag = load_frag(Ab, mrow, kh);
#pragma unroll
    for (int ns = 0; ns < 4; ++ns) {
      int nrow = ns * 16 + (lane & 15);
      v16bf b1 = load_frag(B1b, nrow, kh);
      acc1[ns] = __builtin_amdgcn_wmma_f32_16x16x32_bf16(
          false, afrag, false, b1, (short)0, acc1[ns], false, false);
      v16bf b2 = load_frag(B2b, nrow, kh);
      acc2[ns] = __builtin_amdgcn_wmma_f32_16x16x32_bf16(
          false, afrag, false, b2, (short)0, acc2[ns], false, false);
    }
  }

  // ---- epilogue: scale by gamma(h), write f32
#pragma unroll
  for (int ns = 0; ns < 4; ++ns) {
    int h = n0 + ns * 16 + (lane & 15);
    float gm = gamma_[h];
#pragma unroll
    for (int v = 0; v < 8; ++v) {
      int m = m0 + wave * 16 + v + 8 * (lane >> 4);  // C/D VGPR layout
      bx[(size_t)m * Hh + h] = acc1[ns][v] * gm;
      by[(size_t)m * Hh + h] = acc2[ns][v] * gm;
    }
  }
}

// ---------------------------------------------------------------- segmented rotation scan + SiLU
__global__ void scan_kernel(const float* __restrict__ bx,
                            const float* __restrict__ by,
                            const float* __restrict__ g_,
                            const float* __restrict__ phi_,
                            const float* __restrict__ hc1,
                            const float* __restrict__ hc2,
                            const unsigned char* __restrict__ resets,
                            float* __restrict__ y,
                            float* __restrict__ c1out,
                            float* __restrict__ c2out) {
  int id = blockIdx.x * blockDim.x + threadIdx.x;   // B*H = 4096
  if (id >= Bb * Hh) return;
  int h = id & (Hh - 1);
  int b = id >> 9;

  const float g  = g_[h];
  const float ph = phi_[h];

  float cx, cy;
  {
    float bx0 = bx[(size_t)(b * Tt) * Hh + h];
    float by0 = by[(size_t)(b * Tt) * Hh + h];
    if (resets[b * Tt] == 0) {
      float h1 = hc1[b * Hh + h];
      float h2 = hc2[b * Hh + h];
      bx0 += g * h1 - ph * h2;
      by0 += ph * h1 + g * h2;
    }
    cx = bx0; cy = by0;
    size_t o = (size_t)(b * Tt) * (2 * Hh);
    y[o + h]      = silu_f(cx);
    y[o + Hh + h] = silu_f(cy);
  }

#pragma unroll 4
  for (int t = 1; t < Tt; ++t) {
    size_t src = (size_t)(b * Tt + t) * Hh + h;
    float bxt = bx[src];
    float byt = by[src];
    bool  f   = resets[b * Tt + t] != 0;
    float nx  = f ? bxt : (g * cx - ph * cy + bxt);
    float ny  = f ? byt : (ph * cx + g * cy + byt);
    cx = nx; cy = ny;
    size_t o = (size_t)(b * Tt + t) * (2 * Hh);
    y[o + h]      = silu_f(cx);
    y[o + Hh + h] = silu_f(cy);
  }

  c1out[b * Hh + h] = cx;
  c2out[b * Hh + h] = cy;
}

// ---------------------------------------------------------------- launch
extern "C" void kernel_launch(void* const* d_in, const int* in_sizes, int n_in,
                              void* d_out, int out_size, void* d_ws, size_t ws_size,
                              hipStream_t stream) {
  const float* x         = (const float*)d_in[0];
  const float* nu_log    = (const float*)d_in[1];
  const float* theta_log = (const float*)d_in[2];
  const float* U1        = (const float*)d_in[3];
  const float* U2        = (const float*)d_in[4];
  const float* V1        = (const float*)d_in[5];
  const float* V2        = (const float*)d_in[6];
  const float* hc1       = (const float*)d_in[7];
  const float* hc2       = (const float*)d_in[8];
  const unsigned char* resets = (const unsigned char*)d_in[9];

  char* ws = (char*)d_ws;
  bf16_t* xb  = (bf16_t*)ws;  ws += (size_t)M * Dd * sizeof(bf16_t);
  bf16_t* W1t = (bf16_t*)ws;  ws += (size_t)Hh * Dd * sizeof(bf16_t);
  bf16_t* W2t = (bf16_t*)ws;  ws += (size_t)Hh * Dd * sizeof(bf16_t);
  float*  bx  = (float*)ws;   ws += (size_t)M * Hh * sizeof(float);
  float*  by  = (float*)ws;   ws += (size_t)M * Hh * sizeof(float);
  float*  gp  = (float*)ws;   ws += Hh * sizeof(float);
  float*  php = (float*)ws;   ws += Hh * sizeof(float);
  float*  gmp = (float*)ws;   ws += Hh * sizeof(float);

  float* y     = (float*)d_out;
  float* c1out = y + (size_t)M * 2 * Hh;
  float* c2out = c1out + (size_t)Bb * Hh;

  params_kernel<<<(Hh + 255) / 256, 256, 0, stream>>>(nu_log, theta_log, gp, php, gmp);
  convert_x_kernel<<<(M * Dd / 8) / 256, 256, 0, stream>>>(x, xb);
  build_wt_kernel<<<(Hh * Dd) / 256, 256, 0, stream>>>(U1, U2, V1, V2, W1t, W2t);

  dim3 gg(M / TM, Hh / TN);   // (128, 8)
  gemm_bxby_kernel<<<gg, 256, 0, stream>>>(xb, W1t, W2t, gmp, bx, by);

  scan_kernel<<<(Bb * Hh) / 256, 256, 0, stream>>>(bx, by, gp, php, hc1, hc2,
                                                   resets, y, c1out, c2out);
}
